// MultiHeadAttention_90649579749498
// MI455X (gfx1250) — matrix operand order repaired and checked by reading, not compiled, not run on hardware
//
#include <hip/hip_runtime.h>
#include <hip/hip_bf16.h>
#include <stdint.h>

#define D_MODEL 512
#define SEQ_L   2048
#define BATCH   8
#define TOPK    7

typedef __attribute__((ext_vector_type(16))) __bf16 v16bf;
typedef __attribute__((ext_vector_type(8)))  float  v8f;
typedef int v4i __attribute__((vector_size(16)));  // matches builtin param type

union FragBF { uint4 u[2]; v16bf v; };

#if defined(__HIP_DEVICE_COMPILE__) && \
    __has_builtin(__builtin_amdgcn_global_load_async_to_lds_b128) && \
    __has_builtin(__builtin_amdgcn_s_wait_asynccnt)
#define ASYNC_LDS_OK 1
#else
#define ASYNC_LDS_OK 0
#endif

#define AS1 __attribute__((address_space(1)))
#define AS3 __attribute__((address_space(3)))

__device__ __forceinline__ unsigned short f2bf(float x) {
    unsigned int u = __float_as_uint(x);
    unsigned int r = (u + 0x7FFFu + ((u >> 16) & 1u)) >> 16;  // RNE
    return (unsigned short)r;
}
__device__ __forceinline__ float bf2f(unsigned short h) {
    return __uint_as_float(((unsigned int)h) << 16);
}
__device__ __forceinline__ unsigned int pack2(float a, float b) {
    return (unsigned int)f2bf(a) | ((unsigned int)f2bf(b) << 16);
}

// ---------------------------------------------------------------------------
// C[M,N] = A[M,K] * W[K,N].  A: f32 or bf16(ushort). W: f32. C: f32 or bf16.
// Block tile 128x64, BK=32, 256 threads = 8 waves (4 in M x 2 in N),
// each wave computes 2x2 16x16 tiles via v_wmma_f32_16x16x32_bf16.
// ---------------------------------------------------------------------------
template <typename AT, typename OT>
__global__ __launch_bounds__(256) void gemm_bf16_wmma(const AT* __restrict__ A,
                                                      const float* __restrict__ W,
                                                      OT* __restrict__ C,
                                                      int M, int N, int K) {
    __shared__ __align__(16) unsigned short As[128][40];  // [m][k], pad to 40
    __shared__ __align__(16) unsigned short Bs[64][40];   // [n][k] (W transposed)

    const int tid = threadIdx.x;
    const int m0 = blockIdx.y * 128;
    const int n0 = blockIdx.x * 64;

    const int lane = tid & 31;
    const int lm = lane & 15;
    const int lh = lane >> 4;
    const int waveM = (tid >> 5) >> 1;  // 0..3
    const int waveN = (tid >> 5) & 1;   // 0..1

    v8f acc[2][2];
#pragma unroll
    for (int i = 0; i < 2; ++i)
#pragma unroll
        for (int j = 0; j < 2; ++j)
            acc[i][j] = (v8f){0.f, 0.f, 0.f, 0.f, 0.f, 0.f, 0.f, 0.f};

    for (int kk = 0; kk < K; kk += 32) {
        {  // stage A tile (128 x 32) as bf16
            const int r = tid >> 1;
            const int cb = (tid & 1) * 16;
            if constexpr (sizeof(AT) == 4) {
                const float* ap = (const float*)A + (size_t)(m0 + r) * K + kk + cb;
                float4 f0 = *(const float4*)(ap + 0);
                float4 f1 = *(const float4*)(ap + 4);
                float4 f2 = *(const float4*)(ap + 8);
                float4 f3 = *(const float4*)(ap + 12);
                uint4 u0, u1;
                u0.x = pack2(f0.x, f0.y); u0.y = pack2(f0.z, f0.w);
                u0.z = pack2(f1.x, f1.y); u0.w = pack2(f1.z, f1.w);
                u1.x = pack2(f2.x, f2.y); u1.y = pack2(f2.z, f2.w);
                u1.z = pack2(f3.x, f3.y); u1.w = pack2(f3.z, f3.w);
                *(uint4*)&As[r][cb]     = u0;
                *(uint4*)&As[r][cb + 8] = u1;
            } else {
                const unsigned short* ap =
                    (const unsigned short*)A + (size_t)(m0 + r) * K + kk + cb;
                *(uint4*)&As[r][cb]     = *(const uint4*)(ap + 0);
                *(uint4*)&As[r][cb + 8] = *(const uint4*)(ap + 8);
            }
        }
        {  // stage W tile (32 x 64) transposed -> Bs[n][k]
            const int kr = tid >> 3;       // 0..31
            const int nb = (tid & 7) * 8;  // 0..56
            const float* wp = W + (size_t)(kk + kr) * N + n0 + nb;
            float4 w0 = *(const float4*)(wp + 0);
            float4 w1 = *(const float4*)(wp + 4);
            Bs[nb + 0][kr] = f2bf(w0.x);
            Bs[nb + 1][kr] = f2bf(w0.y);
            Bs[nb + 2][kr] = f2bf(w0.z);
            Bs[nb + 3][kr] = f2bf(w0.w);
            Bs[nb + 4][kr] = f2bf(w1.x);
            Bs[nb + 5][kr] = f2bf(w1.y);
            Bs[nb + 6][kr] = f2bf(w1.z);
            Bs[nb + 7][kr] = f2bf(w1.w);
        }
        __syncthreads();

        FragBF a[2], b[2];
#pragma unroll
        for (int mi = 0; mi < 2; ++mi) {
            const unsigned short* p = &As[waveM * 32 + mi * 16 + lm][lh * 8];
            a[mi].u[0] = *(const uint4*)p;
            a[mi].u[1] = *(const uint4*)(p + 16);
        }
#pragma unroll
        for (int ni = 0; ni < 2; ++ni) {
            const unsigned short* q = &Bs[waveN * 32 + ni * 16 + lm][lh * 8];
            b[ni].u[0] = *(const uint4*)q;
            b[ni].u[1] = *(const uint4*)(q + 16);
        }
#pragma unroll
        for (int mi = 0; mi < 2; ++mi)
#pragma unroll
            for (int ni = 0; ni < 2; ++ni)
                acc[mi][ni] = __builtin_amdgcn_wmma_f32_16x16x32_bf16(
                    false, a[mi].v, false, b[ni].v, (short)0, acc[mi][ni], false, false);
        __syncthreads();
    }

#pragma unroll
    for (int mi = 0; mi < 2; ++mi)
#pragma unroll
        for (int ni = 0; ni < 2; ++ni)
#pragma unroll
            for (int i = 0; i < 8; ++i) {
                int gm = m0 + waveM * 32 + mi * 16 + lh * 8 + i;
                int gn = n0 + waveN * 32 + ni * 16 + lm;
                float v = acc[mi][ni][i];
                if constexpr (sizeof(OT) == 4) C[(size_t)gm * N + gn] = v;
                else                           C[(size_t)gm * N + gn] = f2bf(v);
            }
}

// ---------------------------------------------------------------------------
// mean_value[b,tau] = (1/512) * sum_t <q_s[b,t+tau,:], k_s[b,t,:]>  (circular)
// Score-matrix tiles S = q_s k_s^T reduced onto diagonals tau = (r-c) mod L.
// Block tile 128x128, double-buffered LDS. 8 waves as 2(M) x 4(N); each wave
// computes 4x2 16x16 tiles = 8 WMMAs per K-step. Staging uses CDNA5
// GLOBAL_LOAD_ASYNC_TO_LDS_B128 (ASYNCcnt-pipelined) when available.
// ---------------------------------------------------------------------------
__global__ __launch_bounds__(256) void corr_diag_kernel(const unsigned short* __restrict__ qs,
                                                        const unsigned short* __restrict__ ks,
                                                        float* __restrict__ mean_value) {
    __shared__ __align__(16) unsigned short As[2][128][40];  // [buf][r][k]
    __shared__ __align__(16) unsigned short Bs[2][128][40];  // [buf][c][k]
    __shared__ float mv[256];  // diagonals m-n in [-127,127] -> +127 -> [0,254]

    const int tid = threadIdx.x;
    const int b  = blockIdx.z;
    const int r0 = blockIdx.y * 128;
    const int c0 = blockIdx.x * 128;

    const unsigned short* Aglob = qs + (size_t)b * SEQ_L * D_MODEL;
    const unsigned short* Bglob = ks + (size_t)b * SEQ_L * D_MODEL;

    mv[tid] = 0.0f;

    const int lane = tid & 31;
    const int lm = lane & 15;
    const int lh = lane >> 4;
    const int waveM = (tid >> 5) >> 2;  // 0..1 -> 64 rows each
    const int waveN = (tid >> 5) & 3;   // 0..3 -> 32 cols each

    // per-thread staging coordinates: 128 rows x 32 cols, 2 threads/row
    const int sr = tid >> 1;
    const int sc = (tid & 1) * 16;

    v8f acc[4][2];
#pragma unroll
    for (int i = 0; i < 4; ++i)
#pragma unroll
        for (int j = 0; j < 2; ++j)
            acc[i][j] = (v8f){0.f, 0.f, 0.f, 0.f, 0.f, 0.f, 0.f, 0.f};

    auto stage = [&](int buf, int kk) {
        const unsigned short* ga = Aglob + (size_t)(r0 + sr) * D_MODEL + kk + sc;
        const unsigned short* gb = Bglob + (size_t)(c0 + sr) * D_MODEL + kk + sc;
#if ASYNC_LDS_OK
        __builtin_amdgcn_global_load_async_to_lds_b128(
            (AS1 v4i*)(ga + 0), (AS3 v4i*)&As[buf][sr][sc + 0], 0, 0);
        __builtin_amdgcn_global_load_async_to_lds_b128(
            (AS1 v4i*)(ga + 8), (AS3 v4i*)&As[buf][sr][sc + 8], 0, 0);
        __builtin_amdgcn_global_load_async_to_lds_b128(
            (AS1 v4i*)(gb + 0), (AS3 v4i*)&Bs[buf][sr][sc + 0], 0, 0);
        __builtin_amdgcn_global_load_async_to_lds_b128(
            (AS1 v4i*)(gb + 8), (AS3 v4i*)&Bs[buf][sr][sc + 8], 0, 0);
#else
        uint4 a0 = *(const uint4*)(ga + 0);
        uint4 a1 = *(const uint4*)(ga + 8);
        uint4 b0 = *(const uint4*)(gb + 0);
        uint4 b1 = *(const uint4*)(gb + 8);
        *(uint4*)&As[buf][sr][sc + 0] = a0;
        *(uint4*)&As[buf][sr][sc + 8] = a1;
        *(uint4*)&Bs[buf][sr][sc + 0] = b0;
        *(uint4*)&Bs[buf][sr][sc + 8] = b1;
#endif
    };

    stage(0, 0);

    for (int it = 0; it < D_MODEL / 32; ++it) {
        const int buf = it & 1;
        const bool pref = (it + 1) < (D_MODEL / 32);
        if (pref) stage(buf ^ 1, (it + 1) * 32);
#if ASYNC_LDS_OK
        // Async loads retire in order: waiting to depth 4 guarantees the
        // current buffer's 4 copies are done while the prefetch stays in flight.
        if (pref) __builtin_amdgcn_s_wait_asynccnt(4);
        else      __builtin_amdgcn_s_wait_asynccnt(0);
#endif
        __syncthreads();

        FragBF a[4], bfr[2];
#pragma unroll
        for (int mi = 0; mi < 4; ++mi) {
            const unsigned short* p = &As[buf][waveM * 64 + mi * 16 + lm][lh * 8];
            a[mi].u[0] = *(const uint4*)p;
            a[mi].u[1] = *(const uint4*)(p + 16);
        }
#pragma unroll
        for (int ni = 0; ni < 2; ++ni) {
            const unsigned short* q = &Bs[buf][waveN * 32 + ni * 16 + lm][lh * 8];
            bfr[ni].u[0] = *(const uint4*)q;
            bfr[ni].u[1] = *(const uint4*)(q + 16);
        }
#pragma unroll
        for (int mi = 0; mi < 4; ++mi)
#pragma unroll
            for (int ni = 0; ni < 2; ++ni)
                acc[mi][ni] = __builtin_amdgcn_wmma_f32_16x16x32_bf16(
                    false, a[mi].v, false, bfr[ni].v, (short)0, acc[mi][ni], false, false);
        __syncthreads();
    }

    // reduce 128x128 tile onto its 255 diagonals in LDS
#pragma unroll
    for (int mi = 0; mi < 4; ++mi)
#pragma unroll
        for (int ni = 0; ni < 2; ++ni)
#pragma unroll
            for (int i = 0; i < 8; ++i) {
                int m = waveM * 64 + mi * 16 + lh * 8 + i;
                int n = waveN * 32 + ni * 16 + lm;
                atomicAdd(&mv[m - n + 127], acc[mi][ni][i]);
            }
    __syncthreads();
    if (tid < 255) {
        int tau = r0 - c0 + tid - 127;
        tau = ((tau % SEQ_L) + SEQ_L) % SEQ_L;
        atomicAdd(&mean_value[b * SEQ_L + tau], mv[tid] * (1.0f / 512.0f));
    }
}

__global__ void zero_f32(float* __restrict__ p, int n) {
    int i = blockIdx.x * blockDim.x + threadIdx.x;
    if (i < n) p[i] = 0.0f;
}

// ---------------------------------------------------------------------------
// Top-7 lags of batch-summed mean_value (ranking == mean over batch),
// then per-batch softmax of mean_value[b, idx[k]].
// ---------------------------------------------------------------------------
__global__ __launch_bounds__(256) void topk_softmax_kernel(const float* __restrict__ mean_value,
                                                           int* __restrict__ idx_out,
                                                           float* __restrict__ w_out) {
    __shared__ float s[SEQ_L];
    __shared__ float rv[256];
    __shared__ int   ri[256];
    __shared__ int   tidx[TOPK];
    const int tid = threadIdx.x;
    for (int t = tid; t < SEQ_L; t += 256) {
        float acc = 0.0f;
        for (int b = 0; b < BATCH; ++b) acc += mean_value[b * SEQ_L + t];
        s[t] = acc;
    }
    __syncthreads();
    for (int k = 0; k < TOPK; ++k) {
        float bv = -3.4e38f; int bi = 0;
        for (int t = tid; t < SEQ_L; t += 256) {
            float v = s[t];
            if (v > bv) { bv = v; bi = t; }
        }
        rv[tid] = bv; ri[tid] = bi;
        __syncthreads();
        for (int off = 128; off > 0; off >>= 1) {
            if (tid < off) {
                if (rv[tid + off] > rv[tid] ||
                    (rv[tid + off] == rv[tid] && ri[tid + off] < ri[tid])) {
                    rv[tid] = rv[tid + off];
                    ri[tid] = ri[tid + off];
                }
            }
            __syncthreads();
        }
        if (tid == 0) { tidx[k] = ri[0]; s[ri[0]] = -3.4e38f; }
        __syncthreads();
    }
    if (tid < TOPK) idx_out[tid] = tidx[tid];
    if (tid < BATCH) {
        const int b = tid;
        float w[TOPK]; float m = -3.4e38f;
        for (int k = 0; k < TOPK; ++k) {
            w[k] = mean_value[b * SEQ_L + tidx[k]];
            m = fmaxf(m, w[k]);
        }
        float sum = 0.0f;
        for (int k = 0; k < TOPK; ++k) { w[k] = __expf(w[k] - m); sum += w[k]; }
        for (int k = 0; k < TOPK; ++k) w_out[b * TOPK + k] = w[k] / sum;
    }
}

// ---------------------------------------------------------------------------
// context[b,l,d] = sum_k w[b,k] * v_s[b, (l+idx[k]) % L, d]   (bf16 out)
// ---------------------------------------------------------------------------
__global__ __launch_bounds__(256) void agg_gather_kernel(const unsigned short* __restrict__ vs,
                                                         const int* __restrict__ idx,
                                                         const float* __restrict__ w,
                                                         unsigned short* __restrict__ ctx) {
    const int tid = threadIdx.x;
    const int bid = blockIdx.x;
    const int b = bid / SEQ_L;
    const int l = bid % SEQ_L;
    int id[TOPK]; float wt[TOPK];
#pragma unroll
    for (int k = 0; k < TOPK; ++k) { id[k] = idx[k]; wt[k] = w[b * TOPK + k]; }
    const unsigned short* vb = vs + (size_t)b * SEQ_L * D_MODEL;
    unsigned short* cb = ctx + ((size_t)b * SEQ_L + l) * D_MODEL;
    for (int d = tid; d < D_MODEL; d += 256) {
        float acc = 0.0f;
#pragma unroll
        for (int k = 0; k < TOPK; ++k) {
            int src = (l + id[k]) & (SEQ_L - 1);
            acc += wt[k] * bf2f(vb[(size_t)src * D_MODEL + d]);
        }
        cb[d] = f2bf(acc);
    }
}

extern "C" void kernel_launch(void* const* d_in, const int* in_sizes, int n_in,
                              void* d_out, int out_size, void* d_ws, size_t ws_size,
                              hipStream_t stream) {
    (void)in_sizes; (void)n_in; (void)out_size; (void)ws_size;
    const float* Q   = (const float*)d_in[0];
    const float* K   = (const float*)d_in[1];
    const float* V   = (const float*)d_in[2];
    const float* WQ  = (const float*)d_in[3];
    const float* WK  = (const float*)d_in[4];
    const float* WV  = (const float*)d_in[5];
    const float* Wfc = (const float*)d_in[6];
    float* OUT = (float*)d_out;

    char* ws = (char*)d_ws;
    unsigned short* qs  = (unsigned short*)(ws);                            // 16 MB
    unsigned short* ks  = (unsigned short*)(ws + (size_t)16 * 1024 * 1024); // 16 MB
    unsigned short* vs  = (unsigned short*)(ws + (size_t)32 * 1024 * 1024); // 16 MB
    unsigned short* ctx = (unsigned short*)(ws + (size_t)48 * 1024 * 1024); // 16 MB
    float* mv = (float*)(ws + (size_t)64 * 1024 * 1024);                    // 64 KB
    int*   idx = (int*)(ws + (size_t)64 * 1024 * 1024 + 65536);
    float* tc  = (float*)(ws + (size_t)64 * 1024 * 1024 + 65536 + 256);

    const int Mrows = BATCH * SEQ_L;  // 16384
    dim3 gemmGrid(D_MODEL / 64, Mrows / 128);

    zero_f32<<<64, 256, 0, stream>>>(mv, BATCH * SEQ_L);
    gemm_bf16_wmma<float, unsigned short><<<gemmGrid, 256, 0, stream>>>(Q, WQ, qs, Mrows, D_MODEL, D_MODEL);
    gemm_bf16_wmma<float, unsigned short><<<gemmGrid, 256, 0, stream>>>(K, WK, ks, Mrows, D_MODEL, D_MODEL);
    gemm_bf16_wmma<float, unsigned short><<<gemmGrid, 256, 0, stream>>>(V, WV, vs, Mrows, D_MODEL, D_MODEL);

    dim3 corrGrid(SEQ_L / 128, SEQ_L / 128, BATCH);
    corr_diag_kernel<<<corrGrid, 256, 0, stream>>>(qs, ks, mv);

    topk_softmax_kernel<<<1, 256, 0, stream>>>(mv, idx, tc);
    agg_gather_kernel<<<BATCH * SEQ_L, 256, 0, stream>>>(vs, idx, tc, ctx);

    gemm_bf16_wmma<unsigned short, float><<<gemmGrid, 256, 0, stream>>>(ctx, Wfc, OUT, Mrows, D_MODEL, D_MODEL);
}
